// LSTA_23905787970008
// MI455X (gfx1250) — compile-verified, matrix operand order checked by reading, not run on hardware
//
#include <hip/hip_runtime.h>
#include <hip/hip_bf16.h>

// ---------------------------------------------------------------------------
// MI455X (gfx1250) implementation.
//   B=8, C=768, H=W=96, KT=64, NC=7, NH=8, HD=96, NR=5, NP=4
// Dominant work: 3x3 conv (0.39 TFLOP) -> bf16 WMMA implicit GEMM with the
// input halo staged into LDS via gfx1250 async copies (ASYNCcnt), fused with
// the 1x1 conv + uncertainty epilogue. Deformable-attn value projection is
// computed only at the 640 sampled corners/batch instead of all H*W (~115x fewer MACs).
// ---------------------------------------------------------------------------

#define BB   8
#define CC   768
#define HH   96
#define WW   96
#define HW   (HH*WW)          // 9216
#define KT   64
#define NC   7
#define NH   8
#define HD   96
#define NR   5
#define NPnt 4
#define C1   384              // conv1 out channels
#define KK9  (9*CC)           // 6912 im2col K
#define NWIN 11               // (96-16)/8+1
#define LQ   (BB*NR)          // 40 tokens total
#define APAD 776              // padded channel pitch for the LDS A-tile

typedef __attribute__((ext_vector_type(16))) __bf16 v16bf;
typedef __attribute__((ext_vector_type(8)))  float  v8f;

// ---- bf16 helpers (bit-level RNE, avoids relying on __bf16 scalar ops) -----
__device__ __forceinline__ unsigned short f2bf(float x) {
  union { float f; unsigned int u; } c; c.f = x;
  unsigned int u = c.u;
  u = u + 0x7FFFu + ((u >> 16) & 1u);
  return (unsigned short)(u >> 16);
}
__device__ __forceinline__ v16bf bf16_zero16() {
  union { v16bf v; unsigned int u[8]; } z;
#pragma unroll
  for (int i = 0; i < 8; ++i) z.u[i] = 0u;
  return z.v;
}
__device__ __forceinline__ v16bf load16_f32_to_bf16(const float* p) {
  union { v16bf v; unsigned short s[16]; } u;
#pragma unroll
  for (int i = 0; i < 16; ++i) u.s[i] = f2bf(p[i]);
  return u.v;
}
// 32B bf16 fragment load as two 16B vectors (works for global and LDS, 16B align)
__device__ __forceinline__ v16bf load16u(const unsigned short* p) {
  union { v16bf v; uint4 q[2]; } u;
  u.q[0] = *(const uint4*)(p);
  u.q[1] = *(const uint4*)(p + 8);
  return u.v;
}

// ---------------------------------------------------------------------------
// K0: NCHW f32 -> NHWC bf16 (A-operand layout for the conv implicit GEMM)
// ---------------------------------------------------------------------------
__global__ __launch_bounds__(256) void cvt_feat_kernel(
    const float* __restrict__ f, unsigned short* __restrict__ o) {
  size_t idx = (size_t)blockIdx.x * 256 + threadIdx.x;
  const size_t N = (size_t)BB * HW * CC;
  if (idx >= N) return;
  int c = (int)(idx % CC);
  size_t t = idx / CC;
  int x = (int)(t % WW); t /= WW;
  int y = (int)(t % HH);
  int b = (int)(t / HH);
  o[idx] = f2bf(f[(((size_t)b * CC + c) * HH + y) * WW + x]);
}

// ---------------------------------------------------------------------------
// K1: pack edl_w1 OIHW(384,768,3,3) -> [n=384][k=6912] bf16 ; k=(ky*3+kx)*768+c
// ---------------------------------------------------------------------------
__global__ __launch_bounds__(256) void pack_w1_kernel(
    const float* __restrict__ w, unsigned short* __restrict__ o) {
  size_t idx = (size_t)blockIdx.x * 256 + threadIdx.x;
  const size_t N = (size_t)C1 * KK9;
  if (idx >= N) return;
  int n = (int)(idx / KK9);
  int k = (int)(idx % KK9);
  int blk = k / CC, c = k % CC;
  int ky = blk / 3, kx = blk % 3;
  o[idx] = f2bf(w[(((size_t)n * CC + c) * 3 + ky) * 3 + kx]);
}

// ---------------------------------------------------------------------------
// K2: fused conv3x3(relu) + conv1x1(relu) + uncertainty.
// 1 WG = 16 pixels (one row segment) x 384 channels, 8 waves x 3 N-tiles.
// Phase 1: async-stage the 3x18x768 bf16 input halo into LDS
//          (global_load_async_to_lds_b128, zero-fill at borders).
// Phase 2: K loop = 9 taps x 24 c-steps; A from LDS, B streamed; 3 WMMA/step.
// Phase 3: epilogue reuses the A-tile LDS for hid/evs.
// ---------------------------------------------------------------------------
__global__ __launch_bounds__(256) void conv_fused_kernel(
    const unsigned short* __restrict__ feat,  // (B,H,W,C) bf16
    const unsigned short* __restrict__ w1p,   // (384,6912) bf16
    const float* __restrict__ b1,             // (384)
    const float* __restrict__ w2,             // (7,384)
    const float* __restrict__ b2,             // (7)
    float* __restrict__ unc) {                // (B,H,W)
  __shared__ union SM {
    unsigned short a[3 * 18 * APAD];          // staged halo (bf16)
    struct { float hid[16][C1]; float evs[16][8]; } e;
  } sm;

  const int pix0 = blockIdx.x * 16;
  const int b = pix0 / HW;
  const int rem = pix0 % HW;
  const int y = rem / WW;
  const int x0 = rem % WW;              // 16-pixel tile never crosses a row

  const int tid  = threadIdx.x;
  const int wid  = tid >> 5;
  const int lane = tid & 31;
  const int half = lane >> 4;
  const int l15  = lane & 15;
  const int colbase = wid * 48;         // 3 column tiles per wave

  // ---- Phase 1: stage halo (rows y-1..y+1, pixels x0-1..x0+16) into LDS ----
  // 5184 16-byte chunks: chunk -> (r, px, cb)
  const unsigned atile_lds = (unsigned)(size_t)(void*)sm.a;  // low 32 bits of
                                                             // flat addr = LDS offset
  __builtin_prefetch(w1p + (size_t)((tid * 3) % C1) * KK9, 0, 0);  // warm weights
  for (int chunk = tid; chunk < 3 * 18 * 96; chunk += 256) {
    const int r  = chunk / (18 * 96);
    const int rm = chunk - r * (18 * 96);
    const int px = rm / 96;
    const int cb = rm - px * 96;                       // 16B chunk in pixel
    const int yy = y - 1 + r;
    const int xx = x0 - 1 + px;
    const unsigned ldsOff = (unsigned)(((r * 18 + px) * APAD) * 2 + cb * 16);
    if (yy >= 0 && yy < HH && xx >= 0 && xx < WW) {
      const unsigned long long ga =
          (unsigned long long)(size_t)(feat + (((size_t)b * HH + yy) * WW + xx) * CC)
          + (unsigned)(cb * 16);
      const unsigned ldsAddr = atile_lds + ldsOff;
      asm volatile("global_load_async_to_lds_b128 %0, %1, off"
                   :: "v"(ldsAddr), "v"(ga) : "memory");
    } else {
      uint4 z; z.x = 0u; z.y = 0u; z.z = 0u; z.w = 0u;
      *(uint4*)((char*)sm.a + ldsOff) = z;             // zero-fill border
    }
  }
  asm volatile("s_wait_asynccnt 0x0" ::: "memory");
  __syncthreads();

  // ---- Phase 2: uniform K loop (no divisions, no branches) -----------------
  v8f acc0 = {}; v8f acc1 = {}; v8f acc2 = {};
  for (int blk = 0; blk < 9; ++blk) {
    const int dy = blk / 3, dx = blk % 3;
    const unsigned short* ab = sm.a + ((dy * 18) + l15 + dx) * APAD + half * 16;
    const unsigned short* bb = w1p + (size_t)(colbase + l15) * KK9 + blk * CC + half * 16;
    for (int c0 = 0; c0 < CC; c0 += 32) {
      v16bf a   = load16u(ab + c0);                       // ds_load from staged tile
      v16bf b0  = load16u(bb + c0);
      v16bf b1v = load16u(bb + c0 + (size_t)16 * KK9);
      v16bf b2v = load16u(bb + c0 + (size_t)32 * KK9);
      acc0 = __builtin_amdgcn_wmma_f32_16x16x32_bf16(false, a, false, b0,  (short)0, acc0, false, false);
      acc1 = __builtin_amdgcn_wmma_f32_16x16x32_bf16(false, a, false, b1v, (short)0, acc1, false, false);
      acc2 = __builtin_amdgcn_wmma_f32_16x16x32_bf16(false, a, false, b2v, (short)0, acc2, false, false);
    }
  }
  __syncthreads();                       // all waves done reading the A-tile

  // ---- Phase 3: epilogue (reuses LDS union) --------------------------------
#pragma unroll
  for (int r = 0; r < 8; ++r) {
    const int prow = half * 8 + r;       // pixel index within tile (D layout)
    int c0 = colbase + l15;
    sm.e.hid[prow][c0]      = fmaxf(acc0[r] + b1[c0],      0.f);
    sm.e.hid[prow][c0 + 16] = fmaxf(acc1[r] + b1[c0 + 16], 0.f);
    sm.e.hid[prow][c0 + 32] = fmaxf(acc2[r] + b1[c0 + 32], 0.f);
  }
  __syncthreads();

  if (tid < 16 * NC) {                   // 1x1 conv to 7 classes + relu
    const int p = tid / NC, c = tid % NC;
    float s = b2[c];
    for (int kk = 0; kk < C1; ++kk) s += sm.e.hid[p][kk] * w2[c * C1 + kk];
    sm.e.evs[p][c] = fmaxf(s, 0.f);
  }
  __syncthreads();

  if (tid < 16) {                        // uncertainty = 7 / (sum(ev)+7)
    float s = (float)NC;
    for (int c = 0; c < NC; ++c) s += sm.e.evs[tid][c];
    unc[((size_t)b * HH + y) * WW + x0 + tid] = (float)NC / s;
  }
}

// ---------------------------------------------------------------------------
// K3: 16x16/stride-8 window scores + stable top-5 + box/ref emit (per batch)
// ---------------------------------------------------------------------------
__global__ __launch_bounds__(256) void win_top5_kernel(
    const float* __restrict__ unc, int* __restrict__ rbox,
    int* __restrict__ cbox, float* __restrict__ refp) {
  const int b = blockIdx.x;
  __shared__ float sc[NWIN * NWIN];
  const int tid = threadIdx.x;
  if (tid < NWIN * NWIN) {
    const int i0 = (tid / NWIN) * 8, j0 = (tid % NWIN) * 8;
    float s = 0.f;
    for (int i = 0; i < 16; ++i)
      for (int j = 0; j < 16; ++j)
        s += unc[((size_t)b * HH + i0 + i) * WW + j0 + j];
    sc[tid] = s * (1.0f / 256.0f);
  }
  __syncthreads();
  if (tid == 0) {
    for (int r = 0; r < NR; ++r) {
      int best = 0; float bv = sc[0];
      for (int w = 1; w < NWIN * NWIN; ++w)
        if (sc[w] > bv) { bv = sc[w]; best = w; }   // strict > : stable ties
      sc[best] = -1e30f;
      const int i0 = (best / NWIN) * 8, j0 = (best % NWIN) * 8;
      const int x1 = j0, y1 = i0, x2 = j0 + 16, y2 = i0 + 16;
      int* rb = rbox + (b * NR + r) * 4;
      rb[0] = x1; rb[1] = y1; rb[2] = x2; rb[3] = y2;
      int* cb = cbox + (b * NR + r) * 4;
      cb[0] = (x1 - 5 > 0) ? x1 - 5 : 0;
      cb[1] = (y1 - 5 > 0) ? y1 - 5 : 0;
      cb[2] = (x2 + 5 < WW) ? x2 + 5 : WW;
      cb[3] = (y2 + 5 < HH) ? y2 + 5 : HH;
      refp[(b * NR + r) * 2 + 0] = (float)(x1 + x2) / (2.0f * WW);
      refp[(b * NR + r) * 2 + 1] = (float)(y1 + y2) / (2.0f * HH);
    }
  }
}

// ---------------------------------------------------------------------------
// K4: box-mean token extraction (adaptive_avg_pool to 1x1 == box mean)
// ---------------------------------------------------------------------------
__global__ __launch_bounds__(256) void box_mean_kernel(
    const float* __restrict__ features, const int* __restrict__ boxes,
    float* __restrict__ out) {
  const int br = blockIdx.x;            // b*NR + r
  const int b = br / NR;
  const int* bx = boxes + br * 4;
  const int x1 = bx[0], y1 = bx[1], x2 = bx[2], y2 = bx[3];
  const float inv = 1.0f / (float)((x2 - x1) * (y2 - y1));
  for (int c = threadIdx.x; c < CC; c += 256) {
    const float* fp = features + ((size_t)b * CC + c) * HW;
    float s = 0.f;
    for (int yy = y1; yy < y2; ++yy)
      for (int xx = x1; xx < x2; ++xx) s += fp[yy * WW + xx];
    out[(size_t)br * CC + c] = s * inv;
  }
}

// ---------------------------------------------------------------------------
// K5: generic WMMA GEMM:  C[M,N] = A[M,K] * Bw[N,K]^T (+bias)(+resid)(relu)
// one wave per 16x16 tile; K must be a multiple of 32 (true for all call sites)
// ---------------------------------------------------------------------------
__global__ __launch_bounds__(32) void gemm_wmma_kernel(
    const float* __restrict__ A, const float* __restrict__ Bw,
    const float* __restrict__ bias, const float* __restrict__ resid,
    float* __restrict__ Cm, int M, int N, int K, int relu) {
  const int lane = threadIdx.x & 31;
  const int half = lane >> 4, l15 = lane & 15;
  const int row = blockIdx.x * 16 + l15;   // A row for this lane
  const int col = blockIdx.y * 16 + l15;   // B column (== weight row)
  const bool aok = row < M, bok = col < N;
  v8f acc = {};
  for (int kb = 0; kb < K; kb += 32) {
    const int k = kb + half * 16;
    v16bf a = aok ? load16_f32_to_bf16(A + (size_t)row * K + k) : bf16_zero16();
    v16bf bf = bok ? load16_f32_to_bf16(Bw + (size_t)col * K + k) : bf16_zero16();
    acc = __builtin_amdgcn_wmma_f32_16x16x32_bf16(false, a, false, bf, (short)0, acc, false, false);
  }
#pragma unroll
  for (int r = 0; r < 8; ++r) {
    const int rr = blockIdx.x * 16 + half * 8 + r;
    const int cc = blockIdx.y * 16 + l15;
    if (rr < M && cc < N) {
      float v = acc[r];
      if (bias)  v += bias[cc];
      if (resid) v += resid[(size_t)rr * N + cc];
      if (relu)  v = fmaxf(v, 0.f);
      Cm[(size_t)rr * N + cc] = v;
    }
  }
}

// ---------------------------------------------------------------------------
// K6: LayerNorm over last dim 768 (one block per row)
// ---------------------------------------------------------------------------
__global__ __launch_bounds__(256) void layernorm_kernel(
    const float* __restrict__ in, const float* __restrict__ g,
    const float* __restrict__ bt, float* __restrict__ out) {
  const int row = blockIdx.x;
  __shared__ float red[256];
  const float* x = in + (size_t)row * CC;
  float s = 0.f;
  for (int i = threadIdx.x; i < CC; i += 256) s += x[i];
  red[threadIdx.x] = s; __syncthreads();
  for (int st = 128; st > 0; st >>= 1) {
    if (threadIdx.x < st) red[threadIdx.x] += red[threadIdx.x + st];
    __syncthreads();
  }
  const float mean = red[0] / (float)CC;
  __syncthreads();
  float v = 0.f;
  for (int i = threadIdx.x; i < CC; i += 256) { float d = x[i] - mean; v += d * d; }
  red[threadIdx.x] = v; __syncthreads();
  for (int st = 128; st > 0; st >>= 1) {
    if (threadIdx.x < st) red[threadIdx.x] += red[threadIdx.x + st];
    __syncthreads();
  }
  const float rstd = rsqrtf(red[0] / (float)CC + 1e-5f);
  for (int i = threadIdx.x; i < CC; i += 256)
    out[(size_t)row * CC + i] = (x[i] - mean) * rstd * g[i] + bt[i];
}

// K7: elementwise add
__global__ __launch_bounds__(256) void add_kernel(
    const float* __restrict__ a, const float* __restrict__ b,
    float* __restrict__ o, int n) {
  int i = blockIdx.x * 256 + threadIdx.x;
  if (i < n) o[i] = a[i] + b[i];
}

// ---------------------------------------------------------------------------
// K8: self-attention core on 5 tokens (per (b,h); thread = output dim d)
// qkv layout: (40, 2304) = [q|k|v], head h occupies dims h*96..h*96+95
// ---------------------------------------------------------------------------
__global__ __launch_bounds__(96) void self_attn_kernel(
    const float* __restrict__ qkv, float* __restrict__ outb) {
  const int b = blockIdx.x / NH, h = blockIdx.x % NH;
  const int d = threadIdx.x;
  const float scale = rsqrtf((float)HD);
  float s[NR][NR];
#pragma unroll
  for (int i = 0; i < NR; ++i)
#pragma unroll
    for (int j = 0; j < NR; ++j) {
      float a = 0.f;
      const float* qi = qkv + (size_t)(b * NR + i) * (3 * CC) + h * HD;
      const float* kj = qkv + (size_t)(b * NR + j) * (3 * CC) + CC + h * HD;
      for (int dd = 0; dd < HD; ++dd) a += qi[dd] * kj[dd];
      s[i][j] = a * scale;
    }
#pragma unroll
  for (int i = 0; i < NR; ++i) {
    float m = s[i][0];
    for (int j = 1; j < NR; ++j) m = fmaxf(m, s[i][j]);
    float ss = 0.f;
    for (int j = 0; j < NR; ++j) { s[i][j] = __expf(s[i][j] - m); ss += s[i][j]; }
    const float inv = 1.0f / ss;
    for (int j = 0; j < NR; ++j) s[i][j] *= inv;
  }
#pragma unroll
  for (int i = 0; i < NR; ++i) {
    float o = 0.f;
    for (int j = 0; j < NR; ++j)
      o += s[i][j] * qkv[(size_t)(b * NR + j) * (3 * CC) + 2 * CC + h * HD + d];
    outb[(size_t)(b * NR + i) * CC + h * HD + d] = o;
  }
}

// ---------------------------------------------------------------------------
// K9: deformable sampling. Key optimization: project value ONLY at sampled
// corners (640/batch) instead of all H*W positions (~115x less work).
// grid = B*NR*NH blocks, 96 threads (thread = head dim d).
// ---------------------------------------------------------------------------
__global__ __launch_bounds__(96) void deform_sample_kernel(
    const float* __restrict__ features,   // NCHW f32
    const float* __restrict__ offb,       // (40, 64)
    const float* __restrict__ awb,        // (40, 32) logits
    const float* __restrict__ refp,       // (40, 2)
    const float* __restrict__ vw,         // (768, 768) da_value_w
    const float* __restrict__ vbias,      // (768)
    float* __restrict__ outb) {           // (40, 768)
  const int h  = blockIdx.x % NH;
  const int bq = blockIdx.x / NH;
  const int b  = bq / NR;
  const int d  = threadIdx.x;

  float aw[NPnt];
  {
    float m = -1e30f;
#pragma unroll
    for (int p = 0; p < NPnt; ++p) { aw[p] = awb[bq * (NH * NPnt) + h * NPnt + p]; m = fmaxf(m, aw[p]); }
    float ss = 0.f;
#pragma unroll
    for (int p = 0; p < NPnt; ++p) { aw[p] = __expf(aw[p] - m); ss += aw[p]; }
    const float inv = 1.0f / ss;
#pragma unroll
    for (int p = 0; p < NPnt; ++p) aw[p] *= inv;
  }

  const float rx = refp[bq * 2 + 0], ry = refp[bq * 2 + 1];
  const float* wrow = vw + ((size_t)h * HD + d) * CC;
  const float  bias = vbias[h * HD + d];
  float out = 0.f;

  for (int p = 0; p < NPnt; ++p) {
    const float lx = rx + offb[bq * 64 + (h * NPnt + p) * 2 + 0] * (1.0f / WW);
    const float ly = ry + offb[bq * 64 + (h * NPnt + p) * 2 + 1] * (1.0f / HH);
    const float xf = lx * WW - 0.5f, yf = ly * HH - 0.5f;
    const float x0f = floorf(xf), y0f = floorf(yf);
    const float dx = xf - x0f, dy = yf - y0f;
    const int xi = (int)x0f, yi = (int)y0f;
    const float wx[2] = {1.f - dx, dx};
    const float wy[2] = {1.f - dy, dy};
    float s = 0.f;
#pragma unroll
    for (int cy = 0; cy < 2; ++cy)
#pragma unroll
      for (int cx = 0; cx < 2; ++cx) {
        const int yy = yi + cy, xx = xi + cx;
        if (yy >= 0 && yy < HH && xx >= 0 && xx < WW) {
          const float* fp = features + (size_t)b * CC * HW + yy * WW + xx;
          float v = bias;
          for (int c = 0; c < CC; ++c) v += wrow[c] * fp[(size_t)c * HW];
          s += wy[cy] * wx[cx] * v;
        }
      }
    out += aw[p] * s;
  }
  outb[(size_t)bq * CC + h * HD + d] = out;
}

// ---------------------------------------------------------------------------
// K10: temporal cross-attention core (5 queries x 64 keys per (b,h))
// ---------------------------------------------------------------------------
__global__ __launch_bounds__(256) void cross_attn_kernel(
    const float* __restrict__ qb,   // (40, 768)
    const float* __restrict__ kb,   // (512, 768)
    const float* __restrict__ vb,   // (512, 768)
    float* __restrict__ outb) {     // (40, 768)
  const int b = blockIdx.x / NH, h = blockIdx.x % NH;
  __shared__ float sc[NR * KT];
  const float scale = rsqrtf((float)HD);
  for (int s0 = threadIdx.x; s0 < NR * KT; s0 += 256) {
    const int i = s0 / KT, j = s0 % KT;
    const float* qi = qb + (size_t)(b * NR + i) * CC + h * HD;
    const float* kj = kb + ((size_t)b * KT + j) * CC + h * HD;
    float a = 0.f;
    for (int dd = 0; dd < HD; ++dd) a += qi[dd] * kj[dd];
    sc[s0] = a * scale;
  }
  __syncthreads();
  if (threadIdx.x < NR) {
    float* r = sc + threadIdx.x * KT;
    float m = r[0];
    for (int j = 1; j < KT; ++j) m = fmaxf(m, r[j]);
    float ss = 0.f;
    for (int j = 0; j < KT; ++j) { r[j] = __expf(r[j] - m); ss += r[j]; }
    const float inv = 1.0f / ss;
    for (int j = 0; j < KT; ++j) r[j] *= inv;
  }
  __syncthreads();
  for (int o = threadIdx.x; o < NR * HD; o += 256) {
    const int i = o / HD, d = o % HD;
    float a = 0.f;
    for (int j = 0; j < KT; ++j)
      a += sc[i * KT + j] * vb[((size_t)b * KT + j) * CC + h * HD + d];
    outb[(size_t)(b * NR + i) * CC + h * HD + d] = a;
  }
}

// K11: mean over 5 tokens -> (8, 768)
__global__ __launch_bounds__(256) void mean_pool_kernel(
    const float* __restrict__ tok, float* __restrict__ pooled) {
  int idx = blockIdx.x * 256 + threadIdx.x;
  if (idx >= BB * CC) return;
  const int b = idx / CC, c = idx % CC;
  float s = 0.f;
  for (int r = 0; r < NR; ++r) s += tok[(size_t)(b * NR + r) * CC + c];
  pooled[idx] = s * (1.0f / NR);
}

// ---------------------------------------------------------------------------
static inline void launch_gemm(hipStream_t s, const float* A, const float* Bw,
                               const float* bias, const float* resid, float* C,
                               int M, int N, int K, int relu) {
  dim3 g((M + 15) / 16, (N + 15) / 16);
  gemm_wmma_kernel<<<g, 32, 0, s>>>(A, Bw, bias, resid, C, M, N, K, relu);
}

extern "C" void kernel_launch(void* const* d_in, const int* in_sizes, int n_in,
                              void* d_out, int out_size, void* d_ws, size_t ws_size,
                              hipStream_t stream) {
  (void)in_sizes; (void)n_in; (void)out_size; (void)ws_size;
  // ---- inputs (setup_inputs dict order, params flattened in insertion order)
  const float* features   = (const float*)d_in[0];
  const float* temp_feats = (const float*)d_in[1];
  const float* edl_w1 = (const float*)d_in[2];
  const float* edl_b1 = (const float*)d_in[3];
  const float* edl_w2 = (const float*)d_in[4];
  const float* edl_b2 = (const float*)d_in[5];
  const float* proj_w = (const float*)d_in[6];
  const float* proj_b = (const float*)d_in[7];
  const float* rn_g   = (const float*)d_in[8];
  const float* rn_b   = (const float*)d_in[9];
  const float* n1_g   = (const float*)d_in[10];
  const float* n1_b   = (const float*)d_in[11];
  const float* n2_g   = (const float*)d_in[12];
  const float* n2_b   = (const float*)d_in[13];
  const float* n3_g   = (const float*)d_in[14];
  const float* n3_b   = (const float*)d_in[15];
  const float* sa_in_w  = (const float*)d_in[16];
  const float* sa_in_b  = (const float*)d_in[17];
  const float* sa_out_w = (const float*)d_in[18];
  const float* sa_out_b = (const float*)d_in[19];
  const float* da_value_w = (const float*)d_in[20];
  const float* da_value_b = (const float*)d_in[21];
  const float* da_off_w   = (const float*)d_in[22];
  const float* da_off_b   = (const float*)d_in[23];
  const float* da_attn_w  = (const float*)d_in[24];
  const float* da_attn_b  = (const float*)d_in[25];
  const float* da_out_w   = (const float*)d_in[26];
  const float* da_out_b   = (const float*)d_in[27];
  const float* l1_w = (const float*)d_in[28];
  const float* l1_b = (const float*)d_in[29];
  const float* l2_w = (const float*)d_in[30];
  const float* l2_b = (const float*)d_in[31];
  const float* tc_in_w  = (const float*)d_in[32];
  const float* tc_in_b  = (const float*)d_in[33];
  const float* tc_out_w = (const float*)d_in[34];
  const float* tc_out_b = (const float*)d_in[35];
  const float* tcn_g = (const float*)d_in[36];
  const float* tcn_b = (const float*)d_in[37];
  const float* cls_w1 = (const float*)d_in[38];
  const float* cls_b1 = (const float*)d_in[39];
  const float* cls_w2 = (const float*)d_in[40];
  const float* cls_b2 = (const float*)d_in[41];

  // ---- workspace carve (all offsets 256B aligned)
  char* base = (char*)d_ws;
  size_t off = 0;
  auto carve = [&](size_t bytes) -> void* {
    void* p = base + off;
    off += (bytes + 255) & ~(size_t)255;
    return p;
  };
  unsigned short* featbf = (unsigned short*)carve((size_t)BB * HW * CC * 2);
  unsigned short* w1p    = (unsigned short*)carve((size_t)C1 * KK9 * 2);
  float* unc   = (float*)carve((size_t)BB * HW * 4);
  int*   rbox  = (int*)carve(BB * NR * 4 * sizeof(int));
  int*   cbox  = (int*)carve(BB * NR * 4 * sizeof(int));
  float* refp  = (float*)carve(BB * NR * 2 * sizeof(float));
  const size_t TOKB = (size_t)LQ * CC * 4;
  float* tokRawR = (float*)carve(TOKB);
  float* tokRawC = (float*)carve(TOKB);
  float* bufLin  = (float*)carve(TOKB);
  float* tokR    = (float*)carve(TOKB);
  float* tokC    = (float*)carve(TOKB);
  float* tok0    = (float*)carve(TOKB);
  float* tok1    = (float*)carve(TOKB);
  float* tnorm   = (float*)carve(TOKB);
  float* attnb   = (float*)carve(TOKB);
  float* sampb   = (float*)carve(TOKB);
  float* xb      = (float*)carve(TOKB);
  float* tokF    = (float*)carve(TOKB);
  float* qkv     = (float*)carve((size_t)LQ * 3 * CC * 4);
  float* offb    = (float*)carve((size_t)LQ * 64 * 4);
  float* awb     = (float*)carve((size_t)LQ * 32 * 4);
  float* ffn     = (float*)carve((size_t)LQ * 4 * CC * 4);
  float* qb      = (float*)carve(TOKB);
  float* kb      = (float*)carve((size_t)BB * KT * CC * 4);
  float* vb      = (float*)carve((size_t)BB * KT * CC * 4);
  float* pooled  = (float*)carve((size_t)BB * CC * 4);
  float* clsh    = (float*)carve((size_t)BB * C1 * 4);

  // ---- 1. precision/layout conversion for the conv implicit GEMM
  {
    size_t n = (size_t)BB * HW * CC;
    cvt_feat_kernel<<<dim3((unsigned)((n + 255) / 256)), 256, 0, stream>>>(features, featbf);
    size_t m = (size_t)C1 * KK9;
    pack_w1_kernel<<<dim3((unsigned)((m + 255) / 256)), 256, 0, stream>>>(edl_w1, w1p);
  }
  // ---- 2. fused conv3x3+conv1x1+uncertainty (dominant: 0.39 TFLOP on WMMA)
  conv_fused_kernel<<<dim3(BB * HW / 16), 256, 0, stream>>>(
      featbf, w1p, edl_b1, edl_w2, edl_b2, unc);
  // ---- 3. top-5 regions + boxes + reference points
  win_top5_kernel<<<dim3(BB), 256, 0, stream>>>(unc, rbox, cbox, refp);
  // ---- 4. token extraction (box means) + shared projection + LN, then sum
  box_mean_kernel<<<dim3(BB * NR), 256, 0, stream>>>(features, rbox, tokRawR);
  box_mean_kernel<<<dim3(BB * NR), 256, 0, stream>>>(features, cbox, tokRawC);
  launch_gemm(stream, tokRawR, proj_w, proj_b, nullptr, bufLin, LQ, CC, CC, 0);
  layernorm_kernel<<<dim3(LQ), 256, 0, stream>>>(bufLin, rn_g, rn_b, tokR);
  launch_gemm(stream, tokRawC, proj_w, proj_b, nullptr, bufLin, LQ, CC, CC, 0);
  layernorm_kernel<<<dim3(LQ), 256, 0, stream>>>(bufLin, rn_g, rn_b, tokC);
  add_kernel<<<dim3((LQ * CC + 255) / 256), 256, 0, stream>>>(tokR, tokC, tok0, LQ * CC);
  // ---- 5. pre-norm self-attention over 5 tokens
  layernorm_kernel<<<dim3(LQ), 256, 0, stream>>>(tok0, n1_g, n1_b, tnorm);
  launch_gemm(stream, tnorm, sa_in_w, sa_in_b, nullptr, qkv, LQ, 3 * CC, CC, 0);
  self_attn_kernel<<<dim3(BB * NH), 96, 0, stream>>>(qkv, attnb);
  launch_gemm(stream, attnb, sa_out_w, sa_out_b, tok0, tok1, LQ, CC, CC, 0);
  // ---- 6. deformable cross-attention (value projection only at samples)
  layernorm_kernel<<<dim3(LQ), 256, 0, stream>>>(tok1, n2_g, n2_b, tnorm);
  launch_gemm(stream, tnorm, da_off_w, da_off_b, nullptr, offb, LQ, NH * NPnt * 2, CC, 0);
  launch_gemm(stream, tnorm, da_attn_w, da_attn_b, nullptr, awb, LQ, NH * NPnt, CC, 0);
  deform_sample_kernel<<<dim3(BB * NR * NH), 96, 0, stream>>>(
      features, offb, awb, refp, da_value_w, da_value_b, sampb);
  launch_gemm(stream, sampb, da_out_w, da_out_b, tok1, tok0, LQ, CC, CC, 0);
  // ---- 7. FFN
  layernorm_kernel<<<dim3(LQ), 256, 0, stream>>>(tok0, n3_g, n3_b, tnorm);
  launch_gemm(stream, tnorm, l1_w, l1_b, nullptr, ffn, LQ, 4 * CC, CC, 1);
  launch_gemm(stream, ffn, l2_w, l2_b, tok0, tok1, LQ, CC, 4 * CC, 0);
  // ---- 8. temporal cross-attention
  launch_gemm(stream, tok1, tc_in_w, tc_in_b, nullptr, qb, LQ, CC, CC, 0);
  launch_gemm(stream, temp_feats, tc_in_w + (size_t)CC * CC, tc_in_b + CC,
              nullptr, kb, BB * KT, CC, CC, 0);
  launch_gemm(stream, temp_feats, tc_in_w + (size_t)2 * CC * CC, tc_in_b + 2 * CC,
              nullptr, vb, BB * KT, CC, CC, 0);
  cross_attn_kernel<<<dim3(BB * NH), 256, 0, stream>>>(qb, kb, vb, attnb);
  launch_gemm(stream, attnb, tc_out_w, tc_out_b, tok1, xb, LQ, CC, CC, 0);
  layernorm_kernel<<<dim3(LQ), 256, 0, stream>>>(xb, tcn_g, tcn_b, tokF);
  // ---- 9. pool + classifier -> d_out (8,7)
  mean_pool_kernel<<<dim3((BB * CC + 255) / 256), 256, 0, stream>>>(tokF, pooled);
  launch_gemm(stream, pooled, cls_w1, cls_b1, nullptr, clsh, BB, C1, CC, 1);
  launch_gemm(stream, clsh, cls_w2, cls_b2, nullptr, (float*)d_out, BB, NC, C1, 0);
}
